// MultiHeadAttention_6279242187108
// MI455X (gfx1250) — matrix operand order repaired and checked by reading, not compiled, not run on hardware
//
#include <hip/hip_runtime.h>
#include <hip/hip_bf16.h>

// MI455X / gfx1250 fused multi-head attention.
// All GEMMs via v_wmma_f32_16x16x32_bf16 (wave32, 16x16 tiles, K=32).
// Attention K tiles are staged into LDS with double-buffered async
// global->LDS loads (ASYNCcnt); the 402MB weights stream is written with
// nontemporal stores so it does not evict Kp/Vt from the 192MB L2.
//
// Constants: B=2, T=2048, D=768, H=12, Dk=64, BT=4096.
// d_out = [output f32 (2*2048*768)] ++ [weights f32 (2*12*2048*2048)].
//
// Workspace (~28.5 MiB): 4 bf16 weight matrices, Qp/Kp row-major bf16,
// Vt transposed [B,H,64,2048] bf16, Obf bf16.

typedef __attribute__((ext_vector_type(16))) __bf16       bf16x16;
typedef __attribute__((ext_vector_type(8)))  float        f32x8;
typedef __attribute__((ext_vector_type(4)))  unsigned int u32x4;
typedef __attribute__((ext_vector_type(4)))  int          i32x4;

union Frag16 { bf16x16 h; u32x4 q[2]; };

__device__ __forceinline__ unsigned short f2bf(float f) {
  unsigned int u = __float_as_uint(f);
  u += 0x7FFFu + ((u >> 16) & 1u);   // round-to-nearest-even
  return (unsigned short)(u >> 16);
}

__device__ __forceinline__ f32x8 wmma_bf16(const Frag16& a, const Frag16& b, f32x8 c) {
  return __builtin_amdgcn_wmma_f32_16x16x32_bf16(false, a.h, false, b.h,
                                                 (short)0, c, false, false);
}

// ---- async global->LDS b128 (CDNA5 GLOBAL_LOAD_ASYNC_TO_LDS_B128, ASYNCcnt)
#if __has_builtin(__builtin_amdgcn_global_load_async_to_lds_b128)
#define HAS_ASYNC_LDS 1
#else
#define HAS_ASYNC_LDS 0
#endif

__device__ __forceinline__ void async_ld_b128(const void* gp, void* lp) {
#if HAS_ASYNC_LDS
  void* g = const_cast<void*>(gp);
  __builtin_amdgcn_global_load_async_to_lds_b128(
      (__attribute__((address_space(1))) i32x4*)g,
      (__attribute__((address_space(3))) i32x4*)lp, 0, 0);
#else
  *(u32x4*)lp = *(const u32x4*)gp;   // fallback: stage through VGPRs
#endif
}

__device__ __forceinline__ void wait_async_le1() {
#if HAS_ASYNC_LDS
#if __has_builtin(__builtin_amdgcn_s_wait_asynccnt)
  __builtin_amdgcn_s_wait_asynccnt(1);
#else
  asm volatile("s_wait_asynccnt 0x1" ::: "memory");
#endif
#endif
}
__device__ __forceinline__ void wait_async_le0() {
#if HAS_ASYNC_LDS
#if __has_builtin(__builtin_amdgcn_s_wait_asynccnt)
  __builtin_amdgcn_s_wait_asynccnt(0);
#else
  asm volatile("s_wait_asynccnt 0x0" ::: "memory");
#endif
#endif
}

// ---------------------------------------------------------------- convert
__global__ __launch_bounds__(256) void cvt_f32_to_bf16(
    const float* __restrict__ x, unsigned short* __restrict__ y, int n4) {
  int i = blockIdx.x * 256 + threadIdx.x;
  if (i >= n4) return;
  const float* p = x + (size_t)i * 4;
  union { unsigned long long u; unsigned short s[4]; } pk;
  pk.s[0] = f2bf(p[0]); pk.s[1] = f2bf(p[1]);
  pk.s[2] = f2bf(p[2]); pk.s[3] = f2bf(p[3]);
  ((unsigned long long*)y)[i] = pk.u;
}

// ---------------------------------------------------------------- GEMM
// Y[M,768] = A[M,768] @ W^T  (W row-major [768,768] bf16; B-fragment rows of W)
// MODE 0: A=f32,  Y=bf16 row-major [M,768]
// MODE 1: A=f32,  Y=bf16 transposed per head -> Vt[b,h,d,t]
// MODE 2: A=bf16, Y=f32 row-major [M,768]
// Block: 256 thr (8 waves). Tile: 128M x 64N, K-step 32. Wave = 16M x 64N.
template <int MODE>
__global__ __launch_bounds__(256) void gemm_wmma(
    const void* __restrict__ Aptr, const unsigned short* __restrict__ Bw,
    void* __restrict__ Yptr) {
  __shared__ unsigned short As[128 * 32];   // [row][k] bf16
  __shared__ unsigned short Bs[64 * 32];    // [n][k]   bf16

  const int tid   = threadIdx.x;
  const int mBase = blockIdx.x * 128;
  const int nBase = blockIdx.y * 64;
  const int wave  = tid >> 5;
  const int lane  = tid & 31;
  const int ln    = lane & 15;
  const int hi    = lane >> 4;

  f32x8 acc[4];
#pragma unroll
  for (int nt = 0; nt < 4; ++nt) acc[nt] = (f32x8){0.f,0.f,0.f,0.f,0.f,0.f,0.f,0.f};

  for (int k0 = 0; k0 < 768; k0 += 32) {
    { // stage A tile (convert f32->bf16 unless MODE==2)
      const int r = tid >> 1, c0 = (tid & 1) * 16;
      if (MODE == 2) {
        const unsigned short* Ab = (const unsigned short*)Aptr;
        const u32x4* src = (const u32x4*)(Ab + (size_t)(mBase + r) * 768 + k0 + c0);
        *(u32x4*)&As[r * 32 + c0]     = src[0];
        *(u32x4*)&As[r * 32 + c0 + 8] = src[1];
      } else {
        const float* Af = (const float*)Aptr;
        const float* sp = Af + (size_t)(mBase + r) * 768 + k0 + c0;
        union { u32x4 q; unsigned short s[8]; } p0, p1;
#pragma unroll
        for (int i = 0; i < 8; ++i) { p0.s[i] = f2bf(sp[i]); p1.s[i] = f2bf(sp[8 + i]); }
        *(u32x4*)&As[r * 32 + c0]     = p0.q;
        *(u32x4*)&As[r * 32 + c0 + 8] = p1.q;
      }
    }
    { // stage B tile: 64 rows of W (bf16), 32 K each
      const int r = tid >> 2, c0 = (tid & 3) * 8;
      *(u32x4*)&Bs[r * 32 + c0] =
          *(const u32x4*)(Bw + (size_t)(nBase + r) * 768 + k0 + c0);
    }
    __syncthreads();

    Frag16 aF;
    const int Arow = wave * 16 + ln;
    aF.q[0] = *(const u32x4*)&As[Arow * 32 + 16 * hi];
    aF.q[1] = *(const u32x4*)&As[Arow * 32 + 16 * hi + 8];
#pragma unroll
    for (int nt = 0; nt < 4; ++nt) {
      Frag16 bF;
      const int br = nt * 16 + ln;
      bF.q[0] = *(const u32x4*)&Bs[br * 32 + 16 * hi];
      bF.q[1] = *(const u32x4*)&Bs[br * 32 + 16 * hi + 8];
      acc[nt] = wmma_bf16(aF, bF, acc[nt]);
    }
    __syncthreads();
  }

#pragma unroll
  for (int nt = 0; nt < 4; ++nt) {
#pragma unroll
    for (int v = 0; v < 8; ++v) {
      const int row = mBase + wave * 16 + hi * 8 + v;   // D layout: M = v + 8*hi
      const int col = nBase + nt * 16 + ln;             // N = lane%16
      const float val = acc[nt][v];
      if (MODE == 0) {
        ((unsigned short*)Yptr)[(size_t)row * 768 + col] = f2bf(val);
      } else if (MODE == 1) {
        const int b = row >> 11, t = row & 2047;
        const int h = col >> 6,  d = col & 63;
        ((unsigned short*)Yptr)[(((size_t)b * 12 + h) * 64 + d) * 2048 + t] = f2bf(val);
      } else {
        ((float*)Yptr)[(size_t)row * 768 + col] = val;
      }
    }
  }
}

// ---------------------------------------------------------------- attention
// Block = 128 thr (4 waves), grid = (T/64, B*H). Wave owns 16 query rows.
// K tiles (16 keys x 64 d bf16 = 2KB) are staged into LDS cooperatively with
// double-buffered async global->LDS b128 loads; each thread issues exactly one
// b128 per stage, so "asynccnt<=1" after issuing the next stage completes the
// current one; a block barrier publishes the tile.
// Pass 1: online per-lane softmax stats; 16-lane shfl merge.
// Pass 2: recompute S, nontemporal-write normalized weights (402MB stream,
//         bypass L2), stage P in per-wave LDS, accumulate O += P@V via WMMA.
__global__ __launch_bounds__(128) void attn_fused(
    const unsigned short* __restrict__ Qp, const unsigned short* __restrict__ Kp,
    const unsigned short* __restrict__ Vt, float* __restrict__ Wout,
    unsigned short* __restrict__ Obf) {
  __shared__ unsigned short Qs[64 * 64];      // Q tile [qrow][d]        8KB
  __shared__ unsigned short Ks[2][16 * 64];   // K tile double buffer    4KB
  __shared__ unsigned short Ps[4][16 * 32];   // per-wave P stage        4KB

  const int tid   = threadIdx.x;
  const int qblk  = blockIdx.x;               // 0..31
  const int bh    = blockIdx.y;               // 0..23
  const int b     = bh / 12;
  const int h     = bh - b * 12;
  const int qbase = qblk * 64;
  const int wave  = tid >> 5;
  const int lane  = tid & 31;
  const int ln    = lane & 15;
  const int hi    = lane >> 4;

  const size_t kHead = ((size_t)b * 2048) * 768 + (size_t)h * 64;
  // this thread's slice of a K-tile stage: one b128 (key = tid/8, chunk = tid%8)
  const int skey = tid >> 3, schunk = tid & 7;
  const unsigned short* kSrcBase = Kp + kHead + (size_t)skey * 768 + schunk * 8;
  unsigned short* kDst0 = &Ks[0][skey * 64 + schunk * 8];
  unsigned short* kDst1 = &Ks[1][skey * 64 + schunk * 8];

  { // stage Q tile 64x64 bf16
    const int r = tid >> 1, c0 = (tid & 1) * 32;
    const u32x4* src = (const u32x4*)(Qp + (size_t)(b * 2048 + qbase + r) * 768 + h * 64 + c0);
    u32x4* dst = (u32x4*)&Qs[r * 64 + c0];
    dst[0] = src[0]; dst[1] = src[1]; dst[2] = src[2]; dst[3] = src[3];
  }
  __syncthreads();

  const int Arow = wave * 16 + ln;
  Frag16 aq[2];                                // Q fragments, reused for all key tiles
#pragma unroll
  for (int s = 0; s < 2; ++s) {
    aq[s].q[0] = *(const u32x4*)&Qs[Arow * 64 + s * 32 + 16 * hi];
    aq[s].q[1] = *(const u32x4*)&Qs[Arow * 64 + s * 32 + 16 * hi + 8];
  }

  float maxl[8], suml[8];
#pragma unroll
  for (int v = 0; v < 8; ++v) { maxl[v] = -3.0e38f; suml[v] = 0.f; }

  // ---- pass 1: statistics
  async_ld_b128(kSrcBase, kDst0);                       // stage tile 0
  for (int kt = 0; kt < 128; ++kt) {
    if (kt + 1 < 128) {                                 // stage next tile
      async_ld_b128(kSrcBase + (size_t)(kt + 1) * 16 * 768, (kt & 1) ? kDst0 : kDst1);
      wait_async_le1();
    } else {
      wait_async_le0();
    }
    __syncthreads();                                    // tile kt visible

    const unsigned short* kb = &Ks[kt & 1][0];
    f32x8 S = (f32x8){0.f,0.f,0.f,0.f,0.f,0.f,0.f,0.f};
#pragma unroll
    for (int s = 0; s < 2; ++s) {
      Frag16 bk;
      bk.q[0] = *(const u32x4*)&kb[ln * 64 + s * 32 + 16 * hi];
      bk.q[1] = *(const u32x4*)&kb[ln * 64 + s * 32 + 16 * hi + 8];
      S = wmma_bf16(aq[s], bk, S);
    }
#pragma unroll
    for (int v = 0; v < 8; ++v) {
      const float sv = S[v] * 0.125f;                   // 1/sqrt(64)
      const float nm = fmaxf(maxl[v], sv);
      suml[v] = suml[v] * __expf(maxl[v] - nm) + __expf(sv - nm);
      maxl[v] = nm;
    }
    __syncthreads();                                    // done reading this buffer
  }

  float rmax[8], rinv[8];
#pragma unroll
  for (int v = 0; v < 8; ++v) {                 // merge across the 16-lane row group
    float m = maxl[v];
#pragma unroll
    for (int o = 1; o < 16; o <<= 1) m = fmaxf(m, __shfl_xor(m, o, 32));
    float c = suml[v] * __expf(maxl[v] - m);
#pragma unroll
    for (int o = 1; o < 16; o <<= 1) c += __shfl_xor(c, o, 32);
    rmax[v] = m;
    rinv[v] = 1.0f / c;
  }

  // ---- pass 2: write weights once (nontemporal) + accumulate O = P @ V
  f32x8 oacc[4];
#pragma unroll
  for (int nt = 0; nt < 4; ++nt) oacc[nt] = (f32x8){0.f,0.f,0.f,0.f,0.f,0.f,0.f,0.f};
  unsigned short* ps = &Ps[wave][0];

  async_ld_b128(kSrcBase, kDst0);                       // stage tile 0
  for (int kt = 0; kt < 128; ++kt) {
    if (kt + 1 < 128) {
      async_ld_b128(kSrcBase + (size_t)(kt + 1) * 16 * 768, (kt & 1) ? kDst0 : kDst1);
      wait_async_le1();
    } else {
      wait_async_le0();
    }
    __syncthreads();

    const unsigned short* kb = &Ks[kt & 1][0];
    f32x8 S = (f32x8){0.f,0.f,0.f,0.f,0.f,0.f,0.f,0.f};
#pragma unroll
    for (int s = 0; s < 2; ++s) {
      Frag16 bk;
      bk.q[0] = *(const u32x4*)&kb[ln * 64 + s * 32 + 16 * hi];
      bk.q[1] = *(const u32x4*)&kb[ln * 64 + s * 32 + 16 * hi + 8];
      S = wmma_bf16(aq[s], bk, S);
    }
#pragma unroll
    for (int v = 0; v < 8; ++v) {
      const float p   = __expf(S[v] * 0.125f - rmax[v]) * rinv[v];
      const int   row = qbase + wave * 16 + hi * 8 + v;
      __builtin_nontemporal_store(p, &Wout[((size_t)bh * 2048 + row) * 2048 + kt * 16 + ln]);
      ps[(hi * 8 + v) * 32 + (kt & 1) * 16 + ln] = f2bf(p);        // stage for P@V
    }
    if (kt & 1) {
      asm volatile("s_wait_dscnt 0" ::: "memory");   // wave-local LDS RAW fence
      Frag16 ap;
      ap.q[0] = *(const u32x4*)&ps[ln * 32 + 16 * hi];
      ap.q[1] = *(const u32x4*)&ps[ln * 32 + 16 * hi + 8];
      const int kk0 = (kt >> 1) * 32;
#pragma unroll
      for (int nt = 0; nt < 4; ++nt) {
        const int d = nt * 16 + ln;
        const unsigned short* vp = Vt + ((size_t)bh * 64 + d) * 2048 + kk0 + 16 * hi;
        Frag16 bv;
        bv.q[0] = *(const u32x4*)vp;
        bv.q[1] = *(const u32x4*)(vp + 8);
        oacc[nt] = wmma_bf16(ap, bv, oacc[nt]);
      }
    }
    __syncthreads();
  }

#pragma unroll
  for (int nt = 0; nt < 4; ++nt) {
#pragma unroll
    for (int v = 0; v < 8; ++v) {
      const int row = qbase + wave * 16 + hi * 8 + v;
      Obf[(size_t)(b * 2048 + row) * 768 + h * 64 + nt * 16 + ln] = f2bf(oacc[nt][v]);
    }
  }
}

// ---------------------------------------------------------------- launcher
extern "C" void kernel_launch(void* const* d_in, const int* in_sizes, int n_in,
                              void* d_out, int out_size, void* d_ws, size_t ws_size,
                              hipStream_t stream) {
  (void)in_sizes; (void)n_in; (void)out_size; (void)ws_size;

  const float* query = (const float*)d_in[0];
  const float* key   = (const float*)d_in[1];
  const float* value = (const float*)d_in[2];
  const float* Wq    = (const float*)d_in[3];
  const float* Wk    = (const float*)d_in[4];
  const float* Wv    = (const float*)d_in[5];
  const float* Wo    = (const float*)d_in[6];

  float* out  = (float*)d_out;
  float* wout = out + (size_t)2 * 2048 * 768;          // attention weights region

  char* ws = (char*)d_ws;
  const size_t WSZ = (size_t)768 * 768 * 2;            // one bf16 weight matrix
  const size_t PSZ = (size_t)4096 * 768 * 2;           // one bf16 activation matrix
  unsigned short* Wq_bf = (unsigned short*)(ws);
  unsigned short* Wk_bf = (unsigned short*)(ws + WSZ);
  unsigned short* Wv_bf = (unsigned short*)(ws + 2 * WSZ);
  unsigned short* Wo_bf = (unsigned short*)(ws + 3 * WSZ);
  unsigned short* Qp    = (unsigned short*)(ws + 4 * WSZ);
  unsigned short* Kp    = (unsigned short*)(ws + 4 * WSZ + PSZ);
  unsigned short* Vt    = (unsigned short*)(ws + 4 * WSZ + 2 * PSZ);
  unsigned short* Obf   = (unsigned short*)(ws + 4 * WSZ + 3 * PSZ);

  // 1) weights f32 -> bf16  (768*768/4 = 147456 quads, 576 blocks of 256)
  cvt_f32_to_bf16<<<576, 256, 0, stream>>>(Wq, Wq_bf, 147456);
  cvt_f32_to_bf16<<<576, 256, 0, stream>>>(Wk, Wk_bf, 147456);
  cvt_f32_to_bf16<<<576, 256, 0, stream>>>(Wv, Wv_bf, 147456);
  cvt_f32_to_bf16<<<576, 256, 0, stream>>>(Wo, Wo_bf, 147456);

  // 2) projections: [4096,768] @ W^T
  dim3 ggrid(32, 12);
  gemm_wmma<0><<<ggrid, 256, 0, stream>>>(query, Wq_bf, Qp);
  gemm_wmma<0><<<ggrid, 256, 0, stream>>>(key,   Wk_bf, Kp);
  gemm_wmma<1><<<ggrid, 256, 0, stream>>>(value, Wv_bf, Vt);   // transposed epilogue

  // 3) fused attention: softmax stats + weights write + P@V
  attn_fused<<<dim3(32, 24), 128, 0, stream>>>(Qp, Kp, Vt, wout, Obf);

  // 4) output projection -> f32 d_out
  gemm_wmma<2><<<ggrid, 256, 0, stream>>>(Obf, Wo_bf, out);
}
